// DeformableLightGRU2d_6468220748309
// MI455X (gfx1250) — matrix operand order repaired and checked by reading, not compiled, not run on hardware
//
#include <hip/hip_runtime.h>
#include <hip/hip_bf16.h>

// ---------------------------------------------------------------------------
// DeformableLightGRU2d for MI455X (gfx1250, wave32, WMMA)
//   B=8, Cin=Cout=192, H=W=128.
//   BOTH 3x3 convs run as implicit GEMMs on the matrix cores:
//     gate conv:      M=131072, N=16 (4 used), K=3456, v_wmma_f32_16x16x32_bf16
//     candidate conv: M=131072, N=192,         K=3456, v_wmma_f32_16x16x32_bf16
//   K-pairs packed as dwords so every WMMA fragment is exactly 2x
//   ds_load_b128; im2col (c,t) decomposition carried incrementally;
//   gathers branchless (clamp + select).
// ---------------------------------------------------------------------------

#define BB    8
#define CIN   192
#define COUT  192
#define CTOT  384
#define HH    128
#define WW    128
#define HW    (HH * WW)          // 16384
#define NPIX  (BB * HW)          // 131072
#define KDIM  (CTOT * 9)         // 3456
#define NSTEP (KDIM / 32)        // 108

#define LDA_U 20                 // dword stride, A tile rows  (128 x 16 kpairs)
#define LDB_U 20                 // dword stride, B tile cols  (<=64 x 16 kpairs)

typedef __attribute__((ext_vector_type(16))) __bf16    v16bf;
typedef __attribute__((ext_vector_type(8)))  float     v8f;
typedef __attribute__((ext_vector_type(4)))  unsigned  uint32x4;

__device__ __forceinline__ unsigned short f2bf(float f) {
    unsigned u = __builtin_bit_cast(unsigned, f);
    unsigned r = u + 0x7FFFu + ((u >> 16) & 1u);   // round-to-nearest-even
    return (unsigned short)(r >> 16);
}

__device__ __forceinline__ float sigmoidf(float x) {
    return 1.0f / (1.0f + __expf(-x));
}

// ---------------------------------------------------------------------------
// Kernel X: x, h (f32) -> bf16 combined[B][384][H][W]  (c<192: x, c>=192: h)
// ---------------------------------------------------------------------------
__global__ __launch_bounds__(256)
void xcvt2_kernel(const float* __restrict__ xin,
                  const float* __restrict__ hin,
                  unsigned short* __restrict__ comb) {
    const size_t q = (size_t)blockIdx.x * 256 + threadIdx.x;   // quad index
    if (q >= (size_t)BB * CTOT * HW / 4) return;
    const size_t f  = q * 4;                                    // comb element
    const int    b  = (int)(f / ((size_t)CTOT * HW));
    const size_t rm = f - (size_t)b * CTOT * HW;
    const int    c  = (int)(rm >> 14);
    const size_t s  = rm & (size_t)(HW - 1);
    const float* src = (c < CIN)
                     ? (xin + ((size_t)(b * CIN + c)          << 14) + s)
                     : (hin + ((size_t)(b * COUT + (c - CIN)) << 14) + s);
    const float4 v = *(const float4*)src;
    uint2 u;
    u.x = (unsigned)f2bf(v.x) | ((unsigned)f2bf(v.y) << 16);
    u.y = (unsigned)f2bf(v.z) | ((unsigned)f2bf(v.w) << 16);
    *(uint2*)(comb + f) = u;
}

// ---------------------------------------------------------------------------
// Kernel W: w_can (OIHW f32) -> BmatP[k2][n] dword packing bf16(k=2k2, 2k2+1).
// ---------------------------------------------------------------------------
__global__ __launch_bounds__(256)
void wprep_kernel(const float* __restrict__ w_can, unsigned* __restrict__ BmatP) {
    const int idx = blockIdx.x * blockDim.x + threadIdx.x;
    if (idx >= (KDIM / 2) * COUT) return;
    const int k2 = idx / COUT;
    const int n  = idx - k2 * COUT;
    const unsigned lo = f2bf(w_can[(size_t)n * KDIM + 2 * k2]);
    const unsigned hi = f2bf(w_can[(size_t)n * KDIM + 2 * k2 + 1]);
    BmatP[idx] = lo | (hi << 16);
}

// Kernel Wg: w_gates (4,384,3,3) -> BgP[k2][16], cols 4..15 zero.
__global__ __launch_bounds__(256)
void wprep_g_kernel(const float* __restrict__ w_gates, unsigned* __restrict__ BgP) {
    const int idx = blockIdx.x * blockDim.x + threadIdx.x;
    if (idx >= (KDIM / 2) * 16) return;
    const int k2 = idx >> 4;
    const int n  = idx & 15;
    unsigned v = 0;
    if (n < 4) {
        const unsigned lo = f2bf(w_gates[(size_t)n * KDIM + 2 * k2]);
        const unsigned hi = f2bf(w_gates[(size_t)n * KDIM + 2 * k2 + 1]);
        v = lo | (hi << 16);
    }
    BgP[idx] = v;
}

// ---------------------------------------------------------------------------
// Shared A-stager machinery (identical tile for gate and candidate GEMMs):
// one block = 128 pixels = one (b,y) image row; 16 k-pairs per 32-K slab.
// Each thread owns 2 k-pairs; (c,t) carried incrementally: k+=32 => t+=5
// (mod 9), c+=3 (+carry).
// ---------------------------------------------------------------------------
struct AStager {
    const unsigned short* comb_b;
    int yRow, kpBase, mlo;
    int cS[2][2], tS[2][2];

    __device__ __forceinline__ void init(const unsigned short* cb, int yR, int tid) {
        comb_b = cb; yRow = yR;
        kpBase = (tid >> 5) * 2;
        mlo    = tid & 31;
        #pragma unroll
        for (int kh = 0; kh < 2; ++kh) {
            const int kE = 2 * (kpBase + kh);
            cS[kh][0] = kE / 9;  tS[kh][0] = kE - 9 * (kE / 9);
            const int kO = kE + 1;
            cS[kh][1] = kO / 9;  tS[kh][1] = kO - 9 * (kO / 9);
        }
    }
    __device__ __forceinline__ void stage(unsigned* AsBuf) {
        #pragma unroll
        for (int kh = 0; kh < 2; ++kh) {
            const int kp = kpBase + kh;
            unsigned short vals[2][4];
            #pragma unroll
            for (int eo = 0; eo < 2; ++eo) {
                const int t  = tS[kh][eo];
                const int ky = (t >= 6) ? 2 : ((t >= 3) ? 1 : 0);
                const int kx = t - 3 * ky - 1;        // -1..1
                const int yy = yRow + ky - 1;
                const bool rowOk = ((unsigned)yy < (unsigned)HH);
                const unsigned short* p =
                    comb_b + (((size_t)cS[kh][eo]) << 14) + (rowOk ? yy : 0) * WW;
                #pragma unroll
                for (int mi = 0; mi < 4; ++mi) {
                    const int xr = mlo + mi * 32 + kx;         // -1..128
                    const int xc = min(max(xr, 0), WW - 1);    // clamped, safe
                    const unsigned short raw = p[xc];          // unconditional
                    const bool ok = rowOk & ((unsigned)xr < (unsigned)WW);
                    vals[eo][mi] = ok ? raw : (unsigned short)0;
                }
                int tn = t + 5;
                int cn = cS[kh][eo] + 3;
                if (tn >= 9) { tn -= 9; cn += 1; }
                tS[kh][eo] = tn; cS[kh][eo] = cn;
            }
            #pragma unroll
            for (int mi = 0; mi < 4; ++mi) {
                const int m = mlo + mi * 32;
                AsBuf[m * LDA_U + kp] =
                    (unsigned)vals[0][mi] | ((unsigned)vals[1][mi] << 16);
            }
        }
    }
};

union Frag { v16bf v; uint32x4 q[2]; };

// ---------------------------------------------------------------------------
// Kernel G: gate conv as WMMA implicit GEMM, N=16 (cols 0..3 used).
//   8 waves: wave w owns rows w*16..w*16+15 (1x1 fragment per K-step).
//   Writes raw cc (no bias) planar: cc_ws[g][NPIX], g = 0..3.
// ---------------------------------------------------------------------------
__global__ __launch_bounds__(256)
void gate_gemm_kernel(const unsigned short* __restrict__ comb,
                      const unsigned* __restrict__ BgP,
                      float* __restrict__ cc_ws) {
    __shared__ __align__(16) unsigned AsU[2][128 * LDA_U];  // 2 x 10240 B
    __shared__ __align__(16) unsigned BsU[2][16 * LDB_U];   // 2 x  1280 B

    const int tid  = threadIdx.x;
    const int lane = tid & 31;
    const int wave = tid >> 5;
    const int lm   = lane & 15;
    const int lh   = lane >> 4;

    const int mBase = blockIdx.x * 128;
    const int b     = mBase >> 14;
    const int yRow  = (mBase & (HW - 1)) >> 7;

    AStager st;
    st.init(comb + (size_t)b * CTOT * HW, yRow, tid);
    const int nB16  = tid & 15;        // B stager: 256 dwords/step, 1/thread
    const int kpB16 = tid >> 4;

    v8f acc = (v8f){0.f, 0.f, 0.f, 0.f, 0.f, 0.f, 0.f, 0.f};

    st.stage(AsU[0]);
    BsU[0][nB16 * LDB_U + kpB16] = BgP[(size_t)kpB16 * 16 + nB16];
    __syncthreads();

    for (int step = 0; step < NSTEP; ++step) {
        const int cur = step & 1;
        if (step + 1 < NSTEP) {
            st.stage(AsU[cur ^ 1]);
            BsU[cur ^ 1][nB16 * LDB_U + kpB16] =
                BgP[(size_t)((step + 1) * 16 + kpB16) * 16 + nB16];
        }

        Frag af, bfr;
        const unsigned* arow = &AsU[cur][(wave * 16 + lm) * LDA_U];
        af.q[0] = *(const uint32x4*)(arow + 4 * lh);
        af.q[1] = *(const uint32x4*)(arow + 8 + 4 * lh);
        const unsigned* bcol = &BsU[cur][lm * LDB_U];
        bfr.q[0] = *(const uint32x4*)(bcol + 4 * lh);
        bfr.q[1] = *(const uint32x4*)(bcol + 8 + 4 * lh);

        acc = __builtin_amdgcn_wmma_f32_16x16x32_bf16(
            false, af.v, false, bfr.v, (short)0, acc, false, false);

        __syncthreads();
    }

    // lanes with lm<4 hold gate columns; 8 consecutive pixels per lane
    if (lm < 4) {
        const int pixBase = mBase + wave * 16 + 8 * lh;
        float4 r0 = {acc[0], acc[1], acc[2], acc[3]};
        float4 r1 = {acc[4], acc[5], acc[6], acc[7]};
        float* dst = cc_ws + (size_t)lm * NPIX + pixBase;
        *(float4*)(dst)     = r0;
        *(float4*)(dst + 4) = r1;
    }
}

// ---------------------------------------------------------------------------
// Kernel S: sigmoid gates + grid + bilinear sample of h (f32, zeros padding).
//   Writes z (f32), shifted_h (f32 NCHW), and r*shifted_h as bf16 into the
//   upper 192 channels of combined (overwriting bf16(h), already consumed).
// ---------------------------------------------------------------------------
__global__ __launch_bounds__(256)
void sample_kernel(const float* __restrict__ hin,
                   const float* __restrict__ cc_ws,
                   const float* __restrict__ b_gates,
                   float* __restrict__ z_ws,
                   float* __restrict__ sh_ws,
                   unsigned short* __restrict__ comb) {
    const int pixel = blockIdx.x * 256 + threadIdx.x;
    const int b  = pixel >> 14;
    const int p  = pixel & (HW - 1);
    const int py = p >> 7;
    const int px = p & 127;

    const float cc0 = cc_ws[(size_t)0 * NPIX + pixel] + b_gates[0];
    const float cc1 = cc_ws[(size_t)1 * NPIX + pixel] + b_gates[1];
    const float cc2 = cc_ws[(size_t)2 * NPIX + pixel] + b_gates[2];
    const float cc3 = cc_ws[(size_t)3 * NPIX + pixel] + b_gates[3];

    const float r = sigmoidf(cc0);
    const float z = sigmoidf(cc1);

    // grid (faithful to reference, incl. double-normalized base_y)
    const float base_x = (float)px;
    const float base_y = ((float)py / 64.0f - 1.0f) / 64.0f - 1.0f;
    const float gx = (cc2 + base_x + 1.0f) * 0.5f * (float)(WW - 1);
    const float gy = (cc3 + base_y + 1.0f) * 0.5f * (float)(HH - 1);

    const float x0f = floorf(gx), y0f = floorf(gy);
    const float wx1 = gx - x0f, wx0 = 1.0f - wx1;
    const float wy1 = gy - y0f, wy0 = 1.0f - wy1;
    const int ix0 = (int)x0f, ix1 = ix0 + 1;
    const int iy0 = (int)y0f, iy1 = iy0 + 1;
    const bool vx0 = (ix0 >= 0) & (ix0 < WW), vx1 = (ix1 >= 0) & (ix1 < WW);
    const bool vy0 = (iy0 >= 0) & (iy0 < HH), vy1 = (iy1 >= 0) & (iy1 < HH);
    const int cx0 = min(max(ix0, 0), WW - 1), cx1 = min(max(ix1, 0), WW - 1);
    const int cy0 = min(max(iy0, 0), HH - 1), cy1 = min(max(iy1, 0), HH - 1);
    const int off00 = cy0 * WW + cx0, off01 = cy0 * WW + cx1;
    const int off10 = cy1 * WW + cx0, off11 = cy1 * WW + cx1;
    const float w00 = wy0 * wx0 * ((vy0 & vx0) ? 1.f : 0.f);
    const float w01 = wy0 * wx1 * ((vy0 & vx1) ? 1.f : 0.f);
    const float w10 = wy1 * wx0 * ((vy1 & vx0) ? 1.f : 0.f);
    const float w11 = wy1 * wx1 * ((vy1 & vx1) ? 1.f : 0.f);

    const float* hb      = hin  + (size_t)b * COUT * HW;
    float*       shb     = sh_ws + ((size_t)b * COUT << 14) + p;
    unsigned short* rshc = comb + (((size_t)b * CTOT + CIN) << 14) + p;

    for (int c = 0; c < COUT; ++c) {
        const size_t cb = (size_t)c << 14;
        const float sh = w00 * hb[cb + off00] + w01 * hb[cb + off01]
                       + w10 * hb[cb + off10] + w11 * hb[cb + off11];
        shb[cb]  = sh;
        rshc[cb] = f2bf(r * sh);
    }
    z_ws[pixel] = z;
}

// ---------------------------------------------------------------------------
// Kernel C: candidate conv as WMMA implicit GEMM, double-buffered LDS,
//   fused bias + tanh + GRU-combine epilogue. Tile 128(M) x 64(N);
//   wave (wm, wn) owns 32x32 = 2x2 fragments.
// ---------------------------------------------------------------------------
__global__ __launch_bounds__(256)
void cand_gemm_kernel(const unsigned short* __restrict__ comb,
                      const unsigned* __restrict__ BmatP,
                      const float* __restrict__ b_can,
                      const float* __restrict__ z_ws,
                      const float* __restrict__ sh_ws,
                      float* __restrict__ out) {
    __shared__ __align__(16) unsigned AsU[2][128 * LDA_U];  // 2 x 10240 B
    __shared__ __align__(16) unsigned BsU[2][64 * LDB_U];   // 2 x  5120 B

    const int tid  = threadIdx.x;
    const int lane = tid & 31;
    const int wave = tid >> 5;
    const int wm   = (wave & 3) * 32;
    const int wn   = (wave >> 2) * 32;
    const int lm   = lane & 15;
    const int lh   = lane >> 4;

    const int mBase = blockIdx.x * 128;            // one (b, y) image row
    const int nBase = blockIdx.y * 64;
    const int b     = mBase >> 14;
    const int yRow  = (mBase & (HW - 1)) >> 7;

    AStager st;
    st.init(comb + (size_t)b * CTOT * HW, yRow, tid);
    const int nB   = tid & 63;
    const int kpB4 = (tid >> 6) * 4;

    v8f acc[2][2];
    #pragma unroll
    for (int i = 0; i < 2; ++i)
        #pragma unroll
        for (int j = 0; j < 2; ++j)
            acc[i][j] = (v8f){0.f, 0.f, 0.f, 0.f, 0.f, 0.f, 0.f, 0.f};

    auto stage_b = [&](unsigned* BsBuf, int k02) {
        #pragma unroll
        for (int i = 0; i < 4; ++i) {
            const int kp = kpB4 + i;
            BsBuf[nB * LDB_U + kp] = BmatP[(size_t)(k02 + kp) * COUT + nBase + nB];
        }
    };

    st.stage(AsU[0]);
    stage_b(BsU[0], 0);
    __syncthreads();

    for (int step = 0; step < NSTEP; ++step) {
        const int cur = step & 1;
        if (step + 1 < NSTEP) {
            st.stage(AsU[cur ^ 1]);
            stage_b(BsU[cur ^ 1], (step + 1) * 16);
        }

        Frag af[2], bfr[2];
        #pragma unroll
        for (int si = 0; si < 2; ++si) {
            const unsigned* arow = &AsU[cur][(wm + si * 16 + lm) * LDA_U];
            af[si].q[0] = *(const uint32x4*)(arow + 4 * lh);
            af[si].q[1] = *(const uint32x4*)(arow + 8 + 4 * lh);
        }
        #pragma unroll
        for (int sj = 0; sj < 2; ++sj) {
            const unsigned* bcol = &BsU[cur][(wn + sj * 16 + lm) * LDB_U];
            bfr[sj].q[0] = *(const uint32x4*)(bcol + 4 * lh);
            bfr[sj].q[1] = *(const uint32x4*)(bcol + 8 + 4 * lh);
        }
        #pragma unroll
        for (int si = 0; si < 2; ++si)
            #pragma unroll
            for (int sj = 0; sj < 2; ++sj)
                acc[si][sj] = __builtin_amdgcn_wmma_f32_16x16x32_bf16(
                    false, af[si].v, false, bfr[sj].v,
                    (short)0, acc[si][sj], false, false);

        __syncthreads();
    }

    // ---- epilogue: bias + tanh + z*sh + (1-z)*n ; vectorized b128 I/O ----
    #pragma unroll
    for (int si = 0; si < 2; ++si) {
        const int pixBase = mBase + wm + si * 16 + 8 * lh;   // 8 consecutive px
        const int p       = pixBase & (HW - 1);
        const float4 za = *(const float4*)(z_ws + pixBase);
        const float4 zb = *(const float4*)(z_ws + pixBase + 4);
        const float zv[8] = {za.x, za.y, za.z, za.w, zb.x, zb.y, zb.z, zb.w};
        #pragma unroll
        for (int sj = 0; sj < 2; ++sj) {
            const int n      = nBase + wn + sj * 16 + lm;
            const float bias = b_can[n];
            const size_t o   = ((size_t)(b * COUT + n) << 14) + p;
            const float4 s0  = *(const float4*)(sh_ws + o);
            const float4 s1  = *(const float4*)(sh_ws + o + 4);
            const float sv[8] = {s0.x, s0.y, s0.z, s0.w, s1.x, s1.y, s1.z, s1.w};
            float rr[8];
            #pragma unroll
            for (int v = 0; v < 8; ++v) {
                const float nv = tanhf(acc[si][sj][v] + bias);
                rr[v] = zv[v] * sv[v] + (1.0f - zv[v]) * nv;
            }
            float4 r0 = {rr[0], rr[1], rr[2], rr[3]};
            float4 r1 = {rr[4], rr[5], rr[6], rr[7]};
            *(float4*)(out + o)     = r0;
            *(float4*)(out + o + 4) = r1;
        }
    }
}

// ---------------------------------------------------------------------------
// Launch
// ---------------------------------------------------------------------------
extern "C" void kernel_launch(void* const* d_in, const int* in_sizes, int n_in,
                              void* d_out, int out_size, void* d_ws, size_t ws_size,
                              hipStream_t stream) {
    const float* x       = (const float*)d_in[0];
    const float* h       = (const float*)d_in[1];
    const float* w_gates = (const float*)d_in[2];
    const float* b_gates = (const float*)d_in[3];
    const float* w_can   = (const float*)d_in[4];
    const float* b_can   = (const float*)d_in[5];
    float* out = (float*)d_out;

    char* ws = (char*)d_ws;
    const size_t OFF_Z    = 0;                                      // z f32
    const size_t OFF_SH   = OFF_Z    + (size_t)NPIX * 4;            // shifted_h f32
    const size_t OFF_COMB = OFF_SH   + (size_t)NPIX * COUT * 4;     // combined bf16
    const size_t OFF_B    = OFF_COMB + (size_t)BB * CTOT * HW * 2;  // cand weights
    const size_t OFF_BG   = OFF_B    + (size_t)(KDIM / 2) * COUT * 4; // gate weights
    const size_t OFF_CC   = OFF_BG   + (size_t)(KDIM / 2) * 16 * 4;   // gate planes

    float*          z_ws  = (float*)(ws + OFF_Z);
    float*          sh_ws = (float*)(ws + OFF_SH);
    unsigned short* comb  = (unsigned short*)(ws + OFF_COMB);
    unsigned*       BmatP = (unsigned*)(ws + OFF_B);
    unsigned*       BgP   = (unsigned*)(ws + OFF_BG);
    float*          cc_ws = (float*)(ws + OFF_CC);

    const int nquad2 = BB * CTOT * HW / 4;
    xcvt2_kernel<<<(nquad2 + 255) / 256, 256, 0, stream>>>(x, h, comb);
    wprep_kernel<<<((KDIM / 2) * COUT + 255) / 256, 256, 0, stream>>>(w_can, BmatP);
    wprep_g_kernel<<<((KDIM / 2) * 16 + 255) / 256, 256, 0, stream>>>(w_gates, BgP);
    gate_gemm_kernel<<<NPIX / 128, 256, 0, stream>>>(comb, BgP, cc_ws);
    sample_kernel<<<NPIX / 256, 256, 0, stream>>>(h, cc_ws, b_gates,
                                                  z_ws, sh_ws, comb);
    cand_gemm_kernel<<<dim3(NPIX / 128, COUT / 64), 256, 0, stream>>>(
        comb, BmatP, b_can, z_ws, sh_ws, out);
}
